// MultiHeadAttention2D_38474317037788
// MI455X (gfx1250) — matrix operand order repaired and checked by reading, not compiled, not run on hardware
//
#include <hip/hip_runtime.h>

// ---------------------------------------------------------------------------
// MultiHeadAttention (B=4, S=2048, D=512, H=8, DEPTH=64) for gfx1250 (MI455X)
// bf16 WMMA (f32 accumulate) + flash-attention streaming softmax.
// Mask is bit-packed once (64MB int32 -> 2MB bits) to kill the big HBM stream.
// ---------------------------------------------------------------------------

#define Bn 4
#define Sn 2048
#define Dn 512
#define Hn 8
#define DEPTH 64

typedef __attribute__((ext_vector_type(16))) __bf16 v16bf;
typedef __attribute__((ext_vector_type(8)))  float  v8f;

union Frag16 {
    v16bf v;
    unsigned short u[16];
    unsigned int   d[8];
    uint4          q[2];
};

// Native f32 -> bf16 conversions (clang emits v_cvt_pk_bf16_f32).
__device__ __forceinline__ unsigned short f2bf(float f) {
    union { __bf16 b; unsigned short u; } c;
    c.b = (__bf16)f;
    return c.u;
}
__device__ __forceinline__ unsigned int f2bf2(float lo, float hi) {
    union { __bf16 b[2]; unsigned int u; } c;
    c.b[0] = (__bf16)lo;
    c.b[1] = (__bf16)hi;
    return c.u;
}

__device__ __forceinline__ v8f wmma_bf16(const Frag16& a, const Frag16& b, v8f c) {
    // 8 args: (neg_a, A, neg_b, B, c_mod, C, reuse_a, reuse_b)
    return __builtin_amdgcn_wmma_f32_16x16x32_bf16(false, a.v, false, b.v,
                                                   (short)0, c, false, false);
}

// A-fragment K index for packed pair j (elements 2j,2j+1), contiguous order matching B:
// lanes 0-15: V0..V7 -> K=0..15 ; lanes 16-31: K=16..31
__device__ __forceinline__ int a_k0(int j, int kh) {
    return kh * 16 + 2 * j;
}

// ---------------------------------------------------------------------------
// Kernel 0a: weight prep — transpose fp32 [k][n] -> bf16 [n][k] for all 4 weights
// ---------------------------------------------------------------------------
__global__ void wprep_kernel(const float* __restrict__ wq, const float* __restrict__ wk,
                             const float* __restrict__ wv, const float* __restrict__ wo,
                             unsigned short* __restrict__ wt) {
    int idx = blockIdx.x * blockDim.x + threadIdx.x;
    if (idx >= 4 * Dn * Dn) return;
    int p = idx >> 18;                 // / (512*512)
    int e = idx & (Dn * Dn - 1);
    int n = e >> 9;                    // output row (column of original W)
    int k = e & (Dn - 1);
    const float* W = (p == 0) ? wq : (p == 1) ? wk : (p == 2) ? wv : wo;
    wt[idx] = f2bf(W[k * Dn + n]);
}

// ---------------------------------------------------------------------------
// Kernel 0b: mask prep — pack int32 {0,1} mask into bits (wave ballot).
// mbits[B][S][S/32], bit i of word w = mask[.., w*32+i] != 0
// ---------------------------------------------------------------------------
__global__ void mprep_kernel(const int* __restrict__ mask, unsigned int* __restrict__ mbits) {
    size_t idx = (size_t)blockIdx.x * blockDim.x + threadIdx.x;
    int m = mask[idx];
    unsigned long long bal = __ballot(m != 0);
    if ((threadIdx.x & 31) == 0)
        mbits[idx >> 5] = (unsigned int)bal;
}

// ---------------------------------------------------------------------------
// Kernel 1: fused QKV projections.  One wave = one 16x32 output tile
// (A-fragment conversion reused across two 16-wide column tiles).
//   Qh, Kh : bf16 [B,H,S,64]  (Q pre-scaled by 1/sqrt(DEPTH))
//   Vt     : bf16 [B,H,64,S]  (transposed for contiguous P@V B-fragments)
// ---------------------------------------------------------------------------
__global__ void __launch_bounds__(128) proj_qkv_kernel(
    const float* __restrict__ q, const float* __restrict__ k, const float* __restrict__ v,
    const unsigned short* __restrict__ wtq, const unsigned short* __restrict__ wtk,
    const unsigned short* __restrict__ wtv,
    const float* __restrict__ bq, const float* __restrict__ bk, const float* __restrict__ bv,
    unsigned short* __restrict__ Qh, unsigned short* __restrict__ Kh,
    unsigned short* __restrict__ Vt) {

    const int lane = threadIdx.x & 31;
    const int wid  = threadIdx.x >> 5;
    const int ln   = lane & 15;
    const int kh   = lane >> 4;
    const int proj = blockIdx.z;
    const int rowbase = blockIdx.y * 16;
    const int colbase = (blockIdx.x * 4 + wid) * 32;

    const float* X = (proj == 0) ? q : (proj == 1) ? k : v;
    const unsigned short* W = (proj == 0) ? wtq : (proj == 1) ? wtk : wtv;
    const float* bias = (proj == 0) ? bq : (proj == 1) ? bk : bv;

    const float* xrow = X + (size_t)(rowbase + ln) * Dn;
    const unsigned short* wrow0 = W + (size_t)(colbase + ln) * Dn;
    const unsigned short* wrow1 = W + (size_t)(colbase + 16 + ln) * Dn;

    v8f acc0 = {}, acc1 = {};
    for (int kc = 0; kc < Dn; kc += 32) {
        Frag16 a, b0, b1;
#pragma unroll
        for (int j = 0; j < 8; ++j) {
            int k0 = kc + a_k0(j, kh);
            a.d[j] = f2bf2(xrow[k0], xrow[k0 + 1]);
        }
        const uint4* wp0 = (const uint4*)(wrow0 + kc + kh * 16);
        b0.q[0] = wp0[0]; b0.q[1] = wp0[1];
        const uint4* wp1 = (const uint4*)(wrow1 + kc + kh * 16);
        b1.q[0] = wp1[0]; b1.q[1] = wp1[1];
        acc0 = wmma_bf16(a, b0, acc0);
        acc1 = wmma_bf16(a, b1, acc1);
    }

#pragma unroll
    for (int ct = 0; ct < 2; ++ct) {
        const v8f& acc = ct ? acc1 : acc0;
        const int c = colbase + ct * 16 + ln;
        const int h = c >> 6, dd = c & 63;
        const float bias_n = bias[c];
#pragma unroll
        for (int vv = 0; vv < 8; ++vv) {
            int r = rowbase + vv + 8 * kh;       // global row (b*S + s)
            int b = r >> 11, s = r & (Sn - 1);
            float val = acc[vv] + bias_n;
            if (proj == 0) val *= 0.125f;        // 1/sqrt(64) folded into Q
            size_t bh = (size_t)(b * Hn + h);
            if (proj == 2)
                Vt[(bh * DEPTH + dd) * Sn + s] = f2bf(val);
            else
                ((proj == 0) ? Qh : Kh)[(bh * Sn + s) * DEPTH + dd] = f2bf(val);
        }
    }
}

// ---------------------------------------------------------------------------
// Kernel 2: flash attention. One wave = one (b,h, 32-row q-tile), 64 keys/step.
// 32 WMMAs per 64-key step; every K/V fragment feeds 2 WMMAs.
// ---------------------------------------------------------------------------
__global__ void __launch_bounds__(128) flash_attn_kernel(
    const unsigned short* __restrict__ Qh, const unsigned short* __restrict__ Kh,
    const unsigned short* __restrict__ Vt, const unsigned int* __restrict__ mbits,
    unsigned short* __restrict__ Obf) {

    __shared__ unsigned short lds[4][2 * 16 * 64];   // per-wave P staging (4 KB each)

    const int lane = threadIdx.x & 31;
    const int wid  = threadIdx.x >> 5;
    const int ln   = lane & 15;
    const int kh   = lane >> 4;

    const int wg = blockIdx.x * 4 + wid;             // 0 .. 2047
    const int qt = wg & 63;                          // S/32 q-tiles
    const int bh = wg >> 6;                          // b*H + h
    const int b  = bh >> 3;
    const int h  = bh & 7;
    const int qbase = qt * 32;

    // Preload Q A-fragments: 2 q-subtiles x 2 depth-chunks of 32
    Frag16 qf[2][2];
#pragma unroll
    for (int qs = 0; qs < 2; ++qs) {
        const unsigned short* qrow = Qh + ((size_t)bh * Sn + qbase + qs * 16 + ln) * DEPTH;
#pragma unroll
        for (int cc = 0; cc < 2; ++cc)
#pragma unroll
            for (int j = 0; j < 8; ++j)
                qf[qs][cc].d[j] = *(const unsigned int*)(qrow + cc * 32 + a_k0(j, kh));
    }

    v8f oacc[2][4];
    float mrun[2][8], lrun[2][8];
#pragma unroll
    for (int qs = 0; qs < 2; ++qs) {
#pragma unroll
        for (int nd = 0; nd < 4; ++nd) oacc[qs][nd] = (v8f){0.f, 0.f, 0.f, 0.f, 0.f, 0.f, 0.f, 0.f};
#pragma unroll
        for (int i = 0; i < 8; ++i) { mrun[qs][i] = -3.0e38f; lrun[qs][i] = 0.f; }
    }

    const unsigned short* Kbase = Kh + (size_t)bh * Sn * DEPTH;
    const unsigned short* Vbase = Vt + (size_t)bh * DEPTH * Sn;
    const unsigned int* mbase = mbits + ((size_t)b * Sn + qbase) * (Sn / 32);
    unsigned short* pl = lds[wid];

    for (int kb = 0; kb < Sn; kb += 64) {
        if (kb + 64 < Sn) {   // prefetch next tile (64 K rows, 64 V depth-rows)
            __builtin_prefetch(Kbase + (size_t)(kb + 64 + lane) * DEPTH, 0, 3);
            __builtin_prefetch(Kbase + (size_t)(kb + 96 + lane) * DEPTH, 0, 3);
            __builtin_prefetch(Vbase + (size_t)lane * Sn + kb + 64, 0, 3);
            __builtin_prefetch(Vbase + (size_t)(lane + 32) * Sn + kb + 64, 0, 3);
        }
        // ---- scores: S(32x64) = Q(32x64) @ K^T, four 16-key subtiles ----
        v8f sc[2][4];
#pragma unroll
        for (int qs = 0; qs < 2; ++qs)
#pragma unroll
            for (int sub = 0; sub < 4; ++sub)
                sc[qs][sub] = (v8f){0.f, 0.f, 0.f, 0.f, 0.f, 0.f, 0.f, 0.f};
#pragma unroll
        for (int sub = 0; sub < 4; ++sub) {
            const unsigned short* krow = Kbase + (size_t)(kb + sub * 16 + ln) * DEPTH;
            Frag16 kf0, kf1;
            const uint4* kp0 = (const uint4*)(krow + kh * 16);
            kf0.q[0] = kp0[0]; kf0.q[1] = kp0[1];
            const uint4* kp1 = (const uint4*)(krow + 32 + kh * 16);
            kf1.q[0] = kp1[0]; kf1.q[1] = kp1[1];
#pragma unroll
            for (int qs = 0; qs < 2; ++qs) {
                sc[qs][sub] = wmma_bf16(qf[qs][0], kf0, sc[qs][sub]);
                sc[qs][sub] = wmma_bf16(qf[qs][1], kf1, sc[qs][sub]);
            }
        }
        // ---- bit-packed mask + streaming softmax per q-subtile ----
#pragma unroll
        for (int qs = 0; qs < 2; ++qs) {
            float p[4][8], alpha[8];
#pragma unroll
            for (int vv = 0; vv < 8; ++vv) {
                const unsigned int* mp =
                    mbase + (size_t)(qs * 16 + vv + 8 * kh) * (Sn / 32) + (kb >> 5);
                unsigned int mb0 = mp[0], mb1 = mp[1];
                float s0v = fmaf((float)((mb0 >> ln) & 1u),        -1e9f, sc[qs][0][vv]);
                float s1v = fmaf((float)((mb0 >> (ln + 16)) & 1u), -1e9f, sc[qs][1][vv]);
                float s2v = fmaf((float)((mb1 >> ln) & 1u),        -1e9f, sc[qs][2][vv]);
                float s3v = fmaf((float)((mb1 >> (ln + 16)) & 1u), -1e9f, sc[qs][3][vv]);
                float t = fmaxf(fmaxf(s0v, s1v), fmaxf(s2v, s3v));
#pragma unroll
                for (int xm = 1; xm < 16; xm <<= 1) t = fmaxf(t, __shfl_xor(t, xm, 32));
                float mnew = fmaxf(mrun[qs][vv], t);
                float a  = __expf(mrun[qs][vv] - mnew);
                float e0 = __expf(s0v - mnew);
                float e1 = __expf(s1v - mnew);
                float e2 = __expf(s2v - mnew);
                float e3 = __expf(s3v - mnew);
                float rs = (e0 + e1) + (e2 + e3);
#pragma unroll
                for (int xm = 1; xm < 16; xm <<= 1) rs += __shfl_xor(rs, xm, 32);
                lrun[qs][vv] = lrun[qs][vv] * a + rs;
                mrun[qs][vv] = mnew;
                alpha[vv] = a;
                p[0][vv] = e0; p[1][vv] = e1; p[2][vv] = e2; p[3][vv] = e3;
            }
#pragma unroll
            for (int vv = 0; vv < 8; ++vv)
#pragma unroll
                for (int nd = 0; nd < 4; ++nd) oacc[qs][nd][vv] *= alpha[vv];
            // P: C-layout -> LDS (row-major 16x64 bf16 per subtile)
            unsigned short* pq = pl + qs * 1024;
#pragma unroll
            for (int vv = 0; vv < 8; ++vv) {
                int m = vv + 8 * kh;
                pq[m * 64 + ln]      = f2bf(p[0][vv]);
                pq[m * 64 + 16 + ln] = f2bf(p[1][vv]);
                pq[m * 64 + 32 + ln] = f2bf(p[2][vv]);
                pq[m * 64 + 48 + ln] = f2bf(p[3][vv]);
            }
        }
        // ---- P A-fragments from LDS (2 key-chunks of 32) ----
        Frag16 pf[2][2];
#pragma unroll
        for (int qs = 0; qs < 2; ++qs) {
            const unsigned short* prow = pl + qs * 1024 + ln * 64;
#pragma unroll
            for (int cc = 0; cc < 2; ++cc)
#pragma unroll
                for (int j = 0; j < 8; ++j)
                    pf[qs][cc].d[j] = *(const unsigned int*)(prow + cc * 32 + a_k0(j, kh));
        }
        // ---- O(32x64) += P(32x64) @ V(64x64): each V frag feeds 2 WMMAs ----
#pragma unroll
        for (int nd = 0; nd < 4; ++nd) {
#pragma unroll
            for (int cc = 0; cc < 2; ++cc) {
                Frag16 vf;
                const uint4* vp =
                    (const uint4*)(Vbase + (size_t)(nd * 16 + ln) * Sn + kb + cc * 32 + kh * 16);
                vf.q[0] = vp[0];
                vf.q[1] = vp[1];
#pragma unroll
                for (int qs = 0; qs < 2; ++qs)
                    oacc[qs][nd] = wmma_bf16(pf[qs][cc], vf, oacc[qs][nd]);
            }
        }
    }
    // ---- normalize and store O as bf16 [B,S,D] (head-merged) ----
#pragma unroll
    for (int qs = 0; qs < 2; ++qs)
#pragma unroll
        for (int vv = 0; vv < 8; ++vv) {
            float inv = (lrun[qs][vv] > 0.f) ? (1.0f / lrun[qs][vv]) : 0.f;
            int m = qs * 16 + vv + 8 * kh;
            size_t base = ((size_t)b * Sn + qbase + m) * Dn + h * DEPTH;
            Obf[base + 0  + ln] = f2bf(oacc[qs][0][vv] * inv);
            Obf[base + 16 + ln] = f2bf(oacc[qs][1][vv] * inv);
            Obf[base + 32 + ln] = f2bf(oacc[qs][2][vv] * inv);
            Obf[base + 48 + ln] = f2bf(oacc[qs][3][vv] * inv);
        }
}

// ---------------------------------------------------------------------------
// Kernel 3: output projection  out = Obf @ Wo + bo  (fp32 out, 16x32 per wave)
// ---------------------------------------------------------------------------
__global__ void __launch_bounds__(128) oproj_kernel(
    const unsigned short* __restrict__ Obf, const unsigned short* __restrict__ wto,
    const float* __restrict__ bo, float* __restrict__ out) {

    const int lane = threadIdx.x & 31;
    const int wid  = threadIdx.x >> 5;
    const int ln   = lane & 15;
    const int kh   = lane >> 4;
    const int rowbase = blockIdx.y * 16;
    const int colbase = (blockIdx.x * 4 + wid) * 32;

    const unsigned short* xrow  = Obf + (size_t)(rowbase + ln) * Dn;
    const unsigned short* wrow0 = wto + (size_t)(colbase + ln) * Dn;
    const unsigned short* wrow1 = wto + (size_t)(colbase + 16 + ln) * Dn;

    v8f acc0 = {}, acc1 = {};
    for (int kc = 0; kc < Dn; kc += 32) {
        Frag16 a, b0, b1;
#pragma unroll
        for (int j = 0; j < 8; ++j)
            a.d[j] = *(const unsigned int*)(xrow + kc + a_k0(j, kh));
        const uint4* wp0 = (const uint4*)(wrow0 + kc + kh * 16);
        b0.q[0] = wp0[0]; b0.q[1] = wp0[1];
        const uint4* wp1 = (const uint4*)(wrow1 + kc + kh * 16);
        b1.q[0] = wp1[0]; b1.q[1] = wp1[1];
        acc0 = wmma_bf16(a, b0, acc0);
        acc1 = wmma_bf16(a, b1, acc1);
    }
#pragma unroll
    for (int ct = 0; ct < 2; ++ct) {
        const v8f& acc = ct ? acc1 : acc0;
        const float bias_n = bo[colbase + ct * 16 + ln];
#pragma unroll
        for (int vv = 0; vv < 8; ++vv)
            out[(size_t)(rowbase + vv + 8 * kh) * Dn + colbase + ct * 16 + ln] = acc[vv] + bias_n;
    }
}

// ---------------------------------------------------------------------------
extern "C" void kernel_launch(void* const* d_in, const int* in_sizes, int n_in,
                              void* d_out, int out_size, void* d_ws, size_t ws_size,
                              hipStream_t stream) {
    const float* q    = (const float*)d_in[0];
    const float* k    = (const float*)d_in[1];
    const float* v    = (const float*)d_in[2];
    const int*   mask = (const int*)d_in[3];
    const float* wq   = (const float*)d_in[4];
    const float* bq   = (const float*)d_in[5];
    const float* wk   = (const float*)d_in[6];
    const float* bk   = (const float*)d_in[7];
    const float* wv   = (const float*)d_in[8];
    const float* bv   = (const float*)d_in[9];
    const float* wo   = (const float*)d_in[10];
    const float* bo   = (const float*)d_in[11];

    // workspace (bf16 as ushort):
    //   wt: 4x512*512 (2MB) | Qh,Kh: [B,H,S,64] (8MB each) | Vt: [B,H,64,S] (8MB)
    //   Obf: [B,S,D] (8MB) | mbits: [B,S,S/32] (2MB)   -> total 36MB
    const size_t WSZ = (size_t)Dn * Dn;
    const size_t ASZ = (size_t)Bn * Hn * Sn * DEPTH;
    unsigned short* wt  = (unsigned short*)d_ws;
    unsigned short* Qh  = wt + 4 * WSZ;
    unsigned short* Kh  = Qh + ASZ;
    unsigned short* Vt  = Kh + ASZ;
    unsigned short* Obf = Vt + ASZ;
    unsigned int*   mbits = (unsigned int*)(Obf + (size_t)Bn * Sn * Dn);

    wprep_kernel<<<(4 * Dn * Dn + 255) / 256, 256, 0, stream>>>(wq, wk, wv, wo, wt);
    mprep_kernel<<<((size_t)Bn * Sn * Sn) / 256, 256, 0, stream>>>(mask, mbits);

    dim3 pg(Dn / 32 / 4, (Bn * Sn) / 16, 3);       // (4, 512, 3)
    proj_qkv_kernel<<<pg, 128, 0, stream>>>(q, k, v,
                                            wt, wt + WSZ, wt + 2 * WSZ,
                                            bq, bk, bv, Qh, Kh, Vt);

    flash_attn_kernel<<<(Bn * Hn * (Sn / 32)) / 4, 128, 0, stream>>>(Qh, Kh, Vt, mbits, Obf);

    dim3 og(Dn / 32 / 4, (Bn * Sn) / 16);          // (4, 512)
    oproj_kernel<<<og, 128, 0, stream>>>(Obf, wt + 3 * WSZ, bo, (float*)d_out);
}
